// MACE_58566174048400
// MI455X (gfx1250) — matrix operand order repaired and checked by reading, not compiled, not run on hardware
//
#include <hip/hip_runtime.h>
#include <math.h>

// ---------------------------------------------------------------------------
// MACE-style equivariant GNN forward for MI455X (gfx1250, wave32, WMMA).
// Dense contractions: v_wmma_f32_16x16x32_f16 (f16 in, f32 accum), B tiles
// staged through LDS (f16) once per block, A gathered with b128 loads.
// ---------------------------------------------------------------------------

#define N_NODES 16384
#define N_EDGES 65536
#define N_GRAPHS 64
#define N_ELEM 10
#define CH 128
#define NUM_BESSEL 8
#define RPAD 32           // padded radial feature width (K for first GEMM)
#define R_MAX 5.0f

typedef _Float16 f16;
typedef __attribute__((ext_vector_type(16))) _Float16 v16h;
typedef __attribute__((ext_vector_type(8)))  _Float16 v8h;
typedef __attribute__((ext_vector_type(8)))  float    v8f;

__constant__ int c_LOFM[16] = {0,1,1,1,2,2,2,2,2,3,3,3,3,3,3,3};
static const int h_LOFM[16] = {0,1,1,1,2,2,2,2,2,3,3,3,3,3,3,3};

__device__ __forceinline__ float fast_silu(float x)
{
    return x * __builtin_amdgcn_rcpf(1.0f + __expf(-x));
}

// ---------------------------------------------------------------------------
// WMMA GEMM: C[M x N] = act( scale * (A[M x K] @ B[K x N]) + ADD )
// Row-major A (lda), B (ldb), C (ldc). One wave32 -> one 16x32 output tile
// (two 16x16 WMMA tiles sharing the A operand). Block = 32 x 8 waves
// -> 128 rows x 32 cols; all 8 waves share one LDS-staged f16 B tile.
// Requires: K % 32 == 0, M % 128 == 0, N % 32 == 0, lda % 8 == 0.
// ACT: 0 = none, 1 = SiLU.
// ---------------------------------------------------------------------------
template <int ACT, bool HAS_ADD>
__global__ void __launch_bounds__(256)
gemm_wmma(const float* __restrict__ A, const float* __restrict__ B,
          float* __restrict__ C, const float* __restrict__ ADD,
          int M, int N, int K, int lda, int ldb, int ldc, float scale)
{
    // B tile in LDS: [col][k], 32 cols x 32 k, stride 40 halves (80 B) so
    // each 8-half K-run is 16-byte aligned -> ds_load_b128.
    __shared__ __align__(16) _Float16 sB[32 * 40];

    const int lane = threadIdx.x;            // 0..31
    const int wave = threadIdx.y;            // 0..7
    const int tid  = wave * 32 + lane;
    const int col0 = blockIdx.x * 32;
    const int row0 = (blockIdx.y * 8 + wave) * 16;

    const int l15 = lane & 15;
    const int hi  = lane >> 4;

    // A-operand (ISA 7.12.2, 16-bit A 16x32): lane<16 holds row l15,
    // K = {0..7, 16..23}; lane>=16 holds K = {8..15, 24..31}.
    const int akb = hi * 8;
    const float* Arow = A + (size_t)(row0 + l15) * lda + akb;
    // B-operand: lane<16 -> col l15 with K = {0..7, 16..23}; lane>=16 -> K = {8..15, 24..31}.
    const int bkb = hi * 8;

    // Staging map: thread -> (col = tid&31, krows tid>>5 + 8*s)
    const int stc = tid & 31;
    const int stk = tid >> 5;
    const float* Bst = B + (size_t)stk * ldb + col0 + stc;

    v8f acc0 = {}, acc1 = {};
    for (int k0 = 0; k0 < K; k0 += 32) {
        __syncthreads();                       // previous tile fully consumed
#pragma unroll
        for (int s = 0; s < 4; ++s) {          // stage 32x32 B tile, f32->f16
            const int kk = stk + s * 8;
            sB[stc * 40 + kk] = (f16)Bst[(size_t)(k0 + s * 8) * ldb];
        }
        __syncthreads();

        // A gather: 4x aligned float4 -> global_load_b128
        const float4* Ap = reinterpret_cast<const float4*>(Arow + k0);
        const float4 f0 = Ap[0];   // akb+0..3
        const float4 f1 = Ap[1];   // akb+4..7
        const float4 f2 = Ap[4];   // akb+16..19
        const float4 f3 = Ap[5];   // akb+20..23
        v16h a;
        a[0]  = (f16)f0.x; a[1]  = (f16)f0.y; a[2]  = (f16)f0.z; a[3]  = (f16)f0.w;
        a[4]  = (f16)f1.x; a[5]  = (f16)f1.y; a[6]  = (f16)f1.z; a[7]  = (f16)f1.w;
        a[8]  = (f16)f2.x; a[9]  = (f16)f2.y; a[10] = (f16)f2.z; a[11] = (f16)f2.w;
        a[12] = (f16)f3.x; a[13] = (f16)f3.y; a[14] = (f16)f3.z; a[15] = (f16)f3.w;

        // B from LDS: two 16-byte ds loads per operand
        const v8h* q0 = reinterpret_cast<const v8h*>(sB + l15 * 40 + bkb);
        const v8h* q1 = reinterpret_cast<const v8h*>(sB + (l15 + 16) * 40 + bkb);
        const v8h b0l = q0[0], b0h = q0[2];
        const v8h b1l = q1[0], b1h = q1[2];
        v16h b0, b1;
#pragma unroll
        for (int t = 0; t < 8; ++t) {
            b0[t] = b0l[t]; b0[t + 8] = b0h[t];
            b1[t] = b1l[t]; b1[t + 8] = b1h[t];
        }

        acc0 = __builtin_amdgcn_wmma_f32_16x16x32_f16(
                   false, a, false, b0, (short)0, acc0, false, false);
        acc1 = __builtin_amdgcn_wmma_f32_16x16x32_f16(
                   false, a, false, b1, (short)0, acc1, false, false);
    }

    // C layout: lane<16 -> N=l15, VGPR v -> M=v; lane>=16 -> M=8+v.
    const int cn0 = col0 + l15;
    const int mb  = hi * 8;
#pragma unroll
    for (int v = 0; v < 8; ++v) {
        const size_t r = (size_t)(row0 + mb + v) * ldc;
        float x0 = acc0[v] * scale;
        float x1 = acc1[v] * scale;
        if (HAS_ADD) {
            x0 += ADD[r + cn0];
            x1 += ADD[r + cn0 + 16];
        }
        if (ACT == 1) {
            x0 = fast_silu(x0);
            x1 = fast_silu(x1);
        }
        C[r + cn0]      = x0;
        C[r + cn0 + 16] = x1;
    }
}

// ---------------------------------------------------------------------------
// Small support kernels
// ---------------------------------------------------------------------------
__global__ void zero_kernel(float* __restrict__ p, long n)
{
    long i = (long)blockIdx.x * blockDim.x + threadIdx.x;
    if (i < n) p[i] = 0.0f;
}

// Zero-padded copy of the 8x64 first radial weight into a 32x64 buffer so the
// K=32 GEMM's B reads stay in-bounds (rows 8..31 multiply A-zeros anyway).
__global__ void pad_wr0_kernel(const float* __restrict__ Wr0_t, float* __restrict__ Bpad)
{
    int idx = blockIdx.x * blockDim.x + threadIdx.x;   // 0..2047
    if (idx >= RPAD * 64) return;
    int k = idx >> 6;
    Bpad[idx] = (k < NUM_BESSEL) ? Wr0_t[idx] : 0.0f;
}

__global__ void species_kernel(const float* __restrict__ x, int* __restrict__ sp, int n)
{
    int i = blockIdx.x * blockDim.x + threadIdx.x;
    if (i >= n) return;
    int s = 0; float best = -1.0f;
#pragma unroll
    for (int z = 0; z < N_ELEM; ++z) {
        float v = x[(size_t)i * N_ELEM + z];
        if (v > best) { best = v; s = z; }
    }
    sp[i] = s;
}

__global__ void __launch_bounds__(CH)
embed_kernel(const float* __restrict__ Wemb, const int* __restrict__ sp,
             float* __restrict__ nf)
{
    const int n = blockIdx.x, k = threadIdx.x;
    nf[(size_t)n * CH + k] = Wemb[(size_t)sp[n] * CH + k] * 0.316227766016838f; // 1/sqrt(10)
}

// Per-edge geometry: spherical harmonics (16) + Bessel radial embedding,
// zero-padded to RPAD columns so the radial GEMM runs with K=32 unguarded.
__global__ void edge_geom_kernel(const float* __restrict__ pos,
                                 const int* __restrict__ ei,
                                 float* __restrict__ esh,
                                 float* __restrict__ ef)
{
    int e = blockIdx.x * blockDim.x + threadIdx.x;
    if (e >= N_EDGES) return;
    const int j = ei[e];
    const int i = ei[N_EDGES + e];
    const float vx = pos[j*3+0] - pos[i*3+0];
    const float vy = pos[j*3+1] - pos[i*3+1];
    const float vz = pos[j*3+2] - pos[i*3+2];
    const float r  = sqrtf(vx*vx + vy*vy + vz*vz + 1e-12f);
    const float inv = 1.0f / r;
    const float x = vx*inv, y = vy*inv, z = vz*inv;
    const float x2 = x*x, y2 = y*y, z2 = z*z;

    const float s3 = 1.7320508075688772f, s15 = 3.872983346207417f;
    const float s5 = 2.23606797749979f,   s35_8 = 2.091650066335189f;
    const float s105 = 10.246950765959598f, s21_8 = 1.6201851746019651f;
    const float s7 = 2.6457513110645907f;

    float* Y = esh + (size_t)e * 16;
    Y[0]  = 1.0f;
    Y[1]  = s3 * x;  Y[2] = s3 * y;  Y[3] = s3 * z;
    Y[4]  = s15 * x * y;
    Y[5]  = s15 * y * z;
    Y[6]  = 0.5f * s5 * (3.0f * z2 - 1.0f);
    Y[7]  = s15 * x * z;
    Y[8]  = 0.5f * s15 * (x2 - y2);
    Y[9]  = s35_8 * y * (3.0f * x2 - y2);
    Y[10] = s105 * x * y * z;
    Y[11] = s21_8 * y * (5.0f * z2 - 1.0f);
    Y[12] = 0.5f * s7 * z * (5.0f * z2 - 3.0f);
    Y[13] = s21_8 * x * (5.0f * z2 - 1.0f);
    Y[14] = 0.5f * s105 * z * (x2 - y2);
    Y[15] = s35_8 * x * (x2 - 3.0f * y2);

    // cutoff polynomial, p=5:  1 - 21 u^5 + 35 u^6 - 15 u^7  (u < 1)
    const float u  = r * (1.0f / R_MAX);
    const float u2 = u*u, u4 = u2*u2, u5 = u4*u;
    float fcut = 1.0f - 21.0f*u5 + 35.0f*u5*u - 15.0f*u5*u2;
    fcut = (u < 1.0f) ? fcut : 0.0f;
    const float amp = 0.6324555320336759f * inv * fcut;  // sqrt(2/R_MAX)/r * fcut
    float* F = ef + (size_t)e * RPAD;
    const float w = 3.14159265358979323846f * r * (1.0f / R_MAX);
#pragma unroll
    for (int nb = 1; nb <= NUM_BESSEL; ++nb)
        F[nb - 1] = amp * __sinf((float)nb * w);
#pragma unroll
    for (int nb = NUM_BESSEL; nb < RPAD; ++nb)
        F[nb] = 0.0f;
}

// sc[n,l] = (1/sqrt(CH*N_ELEM)) * sum_k nf[n,k] * W_skip[sp[n],k,l]
__global__ void __launch_bounds__(CH)
skip_kernel(const float* __restrict__ nf, const float* __restrict__ Wskip,
            const int* __restrict__ sp, float* __restrict__ sc)
{
    __shared__ float s_nf[CH];
    const int n = blockIdx.x, l = threadIdx.x;
    s_nf[l] = nf[(size_t)n * CH + l];
    __syncthreads();
    const float* W = Wskip + (size_t)sp[n] * CH * CH;
    float acc = 0.0f;
#pragma unroll 8
    for (int k = 0; k < CH; ++k)
        acc += s_nf[k] * W[(size_t)k * CH + l];
    sc[(size_t)n * CH + l] = acc * 0.02795084971874737f;  // 1/sqrt(1280)
}

// Per-edge message + scatter:  msg[i,m,k] += sh[m]*up[j,k]*tpw[L(m),k] / 10
__global__ void __launch_bounds__(256)
msg_kernel(const int* __restrict__ ei, const float* __restrict__ esh,
           const float* __restrict__ up, const float* __restrict__ tpw,
           float* __restrict__ msg)
{
    __shared__ float s_up[CH];
    __shared__ float s_tpw[4 * CH];
    __shared__ float s_sh[16];
    const int e   = blockIdx.x;
    const int tid = threadIdx.x;
    const int j = ei[e];
    const int i = ei[N_EDGES + e];

    if (tid < CH) s_up[tid] = up[(size_t)j * CH + tid];
    if (tid < 16) s_sh[tid] = esh[(size_t)e * 16 + tid];
#pragma unroll
    for (int t = tid; t < 4 * CH; t += 256)
        s_tpw[t] = tpw[(size_t)e * 4 * CH + t];
    __syncthreads();

    float* dst = msg + (size_t)i * 16 * CH;
    __builtin_prefetch(dst + tid, 1, 0);      // global_prefetch_b8
#pragma unroll
    for (int q = 0; q < 8; ++q) {
        const int idx = q * 256 + tid;        // idx = m*128 + k
        const int m = idx >> 7, k = idx & (CH - 1);
        const float val = s_sh[m] * s_up[k] * s_tpw[c_LOFM[m] * CH + k] * 0.1f;
        atomicAdd(dst + idx, val);
    }
}

// Product basis: b = w1*A0 + sum_l w2[l]*inv2[l] + (sum_l w3[l]*inv2[l])*A0
__global__ void __launch_bounds__(CH)
product_kernel(const float* __restrict__ Abuf, const float* __restrict__ Wp1,
               const float* __restrict__ Wp2, const float* __restrict__ Wp3,
               const int* __restrict__ sp, float* __restrict__ bout)
{
    const int n = blockIdx.x, k = threadIdx.x;
    const int z = sp[n];
    const float* An = Abuf + (size_t)n * 16 * CH;
    const float A0 = An[k];
    float inv2[4] = {0.f, 0.f, 0.f, 0.f};
#pragma unroll
    for (int m = 0; m < 16; ++m) {
        const float a = An[m * CH + k];
        inv2[c_LOFM[m]] += a * a;
    }
    const float w1 = Wp1[(size_t)z * CH + k];
    float s2 = 0.f, s3 = 0.f;
#pragma unroll
    for (int l = 0; l < 4; ++l) {
        s2 += Wp2[((size_t)z * 4 + l) * CH + k] * inv2[l];
        s3 += Wp3[((size_t)z * 4 + l) * CH + k] * inv2[l];
    }
    bout[(size_t)n * CH + k] = w1 * A0 + s2 + s3 * A0;
}

__global__ void __launch_bounds__(CH)
readout_accum(const float* __restrict__ nf, const int* __restrict__ batch,
              float* __restrict__ env, float* __restrict__ cnt)
{
    const int n = blockIdx.x, k = threadIdx.x;
    const int b = batch[n];
    atomicAdd(&env[(size_t)b * CH + k], nf[(size_t)n * CH + k]);
    if (k == 0) atomicAdd(&cnt[b], 1.0f);
}

__global__ void __launch_bounds__(CH)
readout_final(const float* __restrict__ env, const float* __restrict__ cnt,
              float* __restrict__ out)
{
    const int b = blockIdx.x, k = threadIdx.x;
    float c = cnt[b];
    c = (c < 1.0f) ? 1.0f : c;
    out[(size_t)b * CH + k] = env[(size_t)b * CH + k] / c;
}

// ---------------------------------------------------------------------------
// Host orchestration
// ---------------------------------------------------------------------------
static inline void launch_gemm(int act, bool hasAdd,
                               const float* A, const float* B, float* C,
                               const float* ADD, int M, int N, int K,
                               int lda, int ldb, int ldc, float scale,
                               hipStream_t stream)
{
    dim3 block(32, 8);
    dim3 grid(N / 32, M / 128);
    if (act == 1)
        gemm_wmma<1, false><<<grid, block, 0, stream>>>(A, B, C, nullptr, M, N, K, lda, ldb, ldc, scale);
    else if (hasAdd)
        gemm_wmma<0, true><<<grid, block, 0, stream>>>(A, B, C, ADD, M, N, K, lda, ldb, ldc, scale);
    else
        gemm_wmma<0, false><<<grid, block, 0, stream>>>(A, B, C, nullptr, M, N, K, lda, ldb, ldc, scale);
}

extern "C" void kernel_launch(void* const* d_in, const int* in_sizes, int n_in,
                              void* d_out, int out_size, void* d_ws, size_t ws_size,
                              hipStream_t stream)
{
    const float* x         = (const float*)d_in[0];
    const float* pos       = (const float*)d_in[1];
    const float* W_embed   = (const float*)d_in[2];
    const float* W_up      = (const float*)d_in[3];
    const float* Wr0       = (const float*)d_in[4];
    const float* Wr1       = (const float*)d_in[5];
    const float* Wr2       = (const float*)d_in[6];
    const float* Wr3       = (const float*)d_in[7];
    const float* W_lin     = (const float*)d_in[8];
    const float* W_skip    = (const float*)d_in[9];
    const float* Wp1       = (const float*)d_in[10];
    const float* Wp2       = (const float*)d_in[11];
    const float* Wp3       = (const float*)d_in[12];
    const float* W_prodlin = (const float*)d_in[13];
    const int*   edge_idx  = (const int*)d_in[14];
    const int*   batch_idx = (const int*)d_in[15];
    float*       out       = (float*)d_out;

    // Scratch carve-out
    char* w = (char*)d_ws;
    auto carve = [&](size_t bytes) -> void* {
        void* p = (void*)w;
        w += (bytes + 255) & ~(size_t)255;
        return p;
    };
    int*   sp     = (int*)  carve((size_t)N_NODES * sizeof(int));
    float* nf     = (float*)carve((size_t)N_NODES * CH * 4);
    float* sc     = (float*)carve((size_t)N_NODES * CH * 4);
    float* up     = (float*)carve((size_t)N_NODES * CH * 4);
    float* bbuf   = (float*)carve((size_t)N_NODES * CH * 4);
    float* esh    = (float*)carve((size_t)N_EDGES * 16 * 4);
    float* ef     = (float*)carve((size_t)N_EDGES * RPAD * 4);
    float* h0     = (float*)carve((size_t)N_EDGES * 64 * 4);
    float* h1     = (float*)carve((size_t)N_EDGES * 64 * 4);
    float* tpw    = (float*)carve((size_t)N_EDGES * 4 * CH * 4);
    float* msg    = (float*)carve((size_t)N_NODES * 16 * CH * 4);
    float* Abuf   = (float*)carve((size_t)N_NODES * 16 * CH * 4);
    float* wr0pad = (float*)carve((size_t)RPAD * 64 * 4);
    float* env    = (float*)carve((size_t)N_GRAPHS * CH * 4);
    float* cnt    = (float*)carve((size_t)N_GRAPHS * 4);

    const float inv_sqrt_ch = 0.08838834764831845f;  // 1/sqrt(128)

    // Static preprocessing
    species_kernel<<<(N_NODES + 255) / 256, 256, 0, stream>>>(x, sp, N_NODES);
    embed_kernel<<<N_NODES, CH, 0, stream>>>(W_embed, sp, nf);
    edge_geom_kernel<<<(N_EDGES + 255) / 256, 256, 0, stream>>>(pos, edge_idx, esh, ef);

    for (int t = 0; t < 2; ++t) {
        const float* Wup_t = W_up      + (size_t)t * CH * CH;
        const float* Wr0_t = Wr0       + (size_t)t * NUM_BESSEL * 64;
        const float* Wr1_t = Wr1       + (size_t)t * 64 * 64;
        const float* Wr2_t = Wr2       + (size_t)t * 64 * 64;
        const float* Wr3_t = Wr3       + (size_t)t * 64 * 4 * CH;
        const float* Wlin_t = W_lin    + (size_t)t * 4 * CH * CH;
        const float* Wskip_t = W_skip  + (size_t)t * N_ELEM * CH * CH;
        const float* Wp1_t = Wp1       + (size_t)t * N_ELEM * CH;
        const float* Wp2_t = Wp2       + (size_t)t * N_ELEM * 4 * CH;
        const float* Wp3_t = Wp3       + (size_t)t * N_ELEM * 4 * CH;
        const float* Wpl_t = W_prodlin + (size_t)t * CH * CH;

        // Skip connection (species-gathered dense) and node up-projection
        skip_kernel<<<N_NODES, CH, 0, stream>>>(nf, Wskip_t, sp, sc);
        launch_gemm(0, false, nf, Wup_t, up, nullptr,
                    N_NODES, CH, CH, CH, CH, CH, inv_sqrt_ch, stream);

        // Radial MLP on edges (first layer zero-padded to K=32 on both sides)
        pad_wr0_kernel<<<(RPAD * 64 + 255) / 256, 256, 0, stream>>>(Wr0_t, wr0pad);
        launch_gemm(1, false, ef, wr0pad, h0, nullptr,
                    N_EDGES, 64, RPAD, RPAD, 64, 64, 0.3535533905932738f, stream);
        launch_gemm(1, false, h0, Wr1_t, h1, nullptr,
                    N_EDGES, 64, 64, 64, 64, 64, 0.125f, stream);
        launch_gemm(1, false, h1, Wr2_t, h0, nullptr,
                    N_EDGES, 64, 64, 64, 64, 64, 0.125f, stream);
        launch_gemm(0, false, h0, Wr3_t, tpw, nullptr,
                    N_EDGES, 4 * CH, 64, 64, 4 * CH, 4 * CH, 0.125f, stream);

        // Message scatter (segment_sum over destination nodes)
        {
            long nmsg = (long)N_NODES * 16 * CH;
            zero_kernel<<<(unsigned)((nmsg + 255) / 256), 256, 0, stream>>>(msg, nmsg);
        }
        msg_kernel<<<N_EDGES, 256, 0, stream>>>(edge_idx, esh, up, tpw, msg);

        // A[:,m,:] = msg[:,m,:] @ W_lin[L(m)] / sqrt(CH)   (16 strided WMMA GEMMs)
        for (int m = 0; m < 16; ++m) {
            launch_gemm(0, false,
                        msg + (size_t)m * CH,
                        Wlin_t + (size_t)h_LOFM[m] * CH * CH,
                        Abuf + (size_t)m * CH, nullptr,
                        N_NODES, CH, CH, 16 * CH, CH, 16 * CH,
                        inv_sqrt_ch, stream);
        }

        // Symmetric product basis + final linear (+skip) -> new node features
        product_kernel<<<N_NODES, CH, 0, stream>>>(Abuf, Wp1_t, Wp2_t, Wp3_t, sp, bbuf);
        launch_gemm(0, true, bbuf, Wpl_t, nf, sc,
                    N_NODES, CH, CH, CH, CH, CH, inv_sqrt_ch, stream);
    }

    // Graph-mean readout
    {
        long nenv = (long)N_GRAPHS * CH + N_GRAPHS;
        zero_kernel<<<(unsigned)((nenv + 255) / 256), 256, 0, stream>>>(env, nenv);
    }
    readout_accum<<<N_NODES, CH, 0, stream>>>(nf, batch_idx, env, cnt);
    readout_final<<<N_GRAPHS, CH, 0, stream>>>(env, cnt, out);
}
